// BiLSTM_CRF_24670292148293
// MI455X (gfx1250) — matrix operand order repaired and checked by reading, not compiled, not run on hardware
//
#include <hip/hip_runtime.h>
#include <hip/hip_bf16.h>
#include <math.h>

// ---------------- problem constants ----------------
#define Vv 50000
#define Ee 256
#define Hh 512
#define Kk 12
#define Bb 32
#define Tt 256
#define Hd 256
#define G4 1024   // 4*Hd
#define STARTT 10
#define STOPP 11
#define NEGV (-10000.0f)

typedef __attribute__((ext_vector_type(16))) _Float16 v16h;
typedef __attribute__((ext_vector_type(8)))  _Float16 v8h;
typedef __attribute__((ext_vector_type(8)))  float    v8f;

// ---------------- WMMA fragment loaders (ISA §7.12.2 layouts, wave32) ------
// A: 16x32 fp16, row-major with leading dim lda.
// lane l in [0,16): rows m0+l ; hi selects K-halves {hi*8..hi*8+7, 16+hi*8..}
__device__ __forceinline__ v16h load_a_frag(const _Float16* A, int lda,
                                            int m0, int k0, int l, int hi) {
  const _Float16* p = A + (size_t)(m0 + l) * lda + k0 + hi * 8;
  v8h lo = *reinterpret_cast<const v8h*>(p);
  v8h hh = *reinterpret_cast<const v8h*>(p + 16);
  return __builtin_shufflevector(lo, hh, 0,1,2,3,4,5,6,7,8,9,10,11,12,13,14,15);
}
// B: 32x16 fp16 where B(k,n) = W[n][k], W row-major [N,Kdim] ld=ldw.
// lane l = column n0+l ; hi selects K-halves {hi*8..hi*8+7, 16+hi*8..}
__device__ __forceinline__ v16h load_b_frag(const _Float16* W, int ldw,
                                            int n0, int k0, int l, int hi) {
  const _Float16* p = W + (size_t)(n0 + l) * ldw + k0 + hi * 8;
  v8h lo = *reinterpret_cast<const v8h*>(p);
  v8h hh = *reinterpret_cast<const v8h*>(p + 16);
  return __builtin_shufflevector(lo, hh, 0,1,2,3,4,5,6,7,8,9,10,11,12,13,14,15);
}

__device__ __forceinline__ float sigf(float x) {
  return 1.0f / (1.0f + __expf(-x));
}

// ---------------- CDNA5 async Global->LDS copy (ASYNCcnt path) -------------
// VDST operand holds the LDS byte address; per ISA §10.2 the LDS-relative
// address is the low 32 bits of a generic shared-space pointer.
__device__ __forceinline__ void async_copy_b128(void* lds_dst, const void* g_src) {
  unsigned int laddr = (unsigned int)(unsigned long long)lds_dst;
  asm volatile("global_load_async_to_lds_b128 %0, %1, off"
               :: "v"(laddr), "v"(g_src) : "memory");
}
__device__ __forceinline__ void wait_asynccnt0() {
  asm volatile("s_wait_asynccnt 0" ::: "memory");
}

// ---------------- kernel 1: weight fp32 -> fp16 ----------------
__global__ void k_convert_w(const float* wih_f, const float* whh_f,
                            const float* wih_b, const float* whh_b,
                            _Float16* o_wih_f, _Float16* o_whh_f,
                            _Float16* o_wih_b, _Float16* o_whh_b) {
  int i = blockIdx.x * blockDim.x + threadIdx.x;  // 0 .. 262143
  const float* src; _Float16* dst;
  switch (blockIdx.y) {
    case 0: src = wih_f; dst = o_wih_f; break;
    case 1: src = whh_f; dst = o_whh_f; break;
    case 2: src = wih_b; dst = o_wih_b; break;
    default: src = whh_b; dst = o_whh_b; break;
  }
  dst[i] = (_Float16)src[i];
}

// ---------------- kernel 2: embedding gather (fwd + per-seq reversed) ------
__global__ void k_embed(const int* sent, const int* lens, const float* emb,
                        _Float16* xf, _Float16* xr) {
  int idx = blockIdx.x * 256 + threadIdx.x;   // < B*T*E
  int e  = idx & (Ee - 1);
  int bt = idx >> 8;
  int t  = bt & (Tt - 1);
  int b  = bt >> 8;
  int s = sent[b * Tt + t];
  xf[idx] = (_Float16)emb[(size_t)s * Ee + e];
  int len = lens[b];
  int t2 = (t < len) ? (len - 1 - t) : t;
  int s2 = sent[b * Tt + t2];
  xr[idx] = (_Float16)emb[(size_t)s2 * Ee + e];
}

// ---------------- kernel 3: input projection GEMM (both dirs) --------------
// G[t][b][n] = sum_k x[b][t][k]*Wih[n][k] + bih[n] + bhh[n]
__global__ void k_gemm_in(const _Float16* xf, const _Float16* xr,
                          const _Float16* wf, const _Float16* wb,
                          const float* bih_f, const float* bhh_f,
                          const float* bih_b, const float* bhh_b,
                          float* Gf, float* Gb) {
  const int dir = blockIdx.y;
  const _Float16* X = dir ? xr : xf;
  const _Float16* W = dir ? wb : wf;
  const float* b1 = dir ? bih_b : bih_f;
  const float* b2 = dir ? bhh_b : bhh_f;
  float* G = dir ? Gb : Gf;

  const int wave = threadIdx.x >> 5;
  const int lane = threadIdx.x & 31;
  const int l = lane & 15, hi = lane >> 4;
  const int tile = blockIdx.x * 8 + wave;     // 0 .. 32767
  const int tn = tile & 63, tm = tile >> 6;   // 64 N-tiles, 512 M-tiles
  const int m0 = tm * 16, n0 = tn * 16;

  v8f acc = {};
#pragma unroll
  for (int k0 = 0; k0 < Ee; k0 += 32) {
    v16h a  = load_a_frag(X, Ee, m0, k0, l, hi);
    v16h bf = load_b_frag(W, Ee, n0, k0, l, hi);
    acc = __builtin_amdgcn_wmma_f32_16x16x32_f16(false, a, false, bf,
                                                 (short)0, acc, false, false);
  }
  const float bias = b1[n0 + l] + b2[n0 + l];
#pragma unroll
  for (int r = 0; r < 8; ++r) {
    int row = m0 + r + hi * 8;       // = b*T + t
    int b_ = row >> 8, t_ = row & 255;
    G[((size_t)t_ * Bb + b_) * G4 + n0 + l] = acc[r] + bias;
  }
}

// ---------------- kernel 4: persistent LSTM recurrence ---------------------
// grid = 2 (direction), block = 1024 (32 waves).
// LDS: h(fp16) 16K | c(f32) 32K | gates(f32) 128K | G-stage(f32) 128K = 304K.
// Next step's gate pre-activations are DMA'd Global->LDS with the CDNA5
// async path (ASYNCcnt) while the current step computes.
__global__ void k_lstm(const _Float16* whh_f, const _Float16* whh_b,
                       const float* Gf, const float* Gb,
                       const float* h0, const float* c0,
                       float* Hf, float* Hb) {
  extern __shared__ __align__(16) unsigned char smem[];
  _Float16* h16  = (_Float16*)smem;                 // 32x256 fp16 : 16 KB
  float*    cst  = (float*)(smem + 16384);          // 32x256 f32  : 32 KB
  float*    gate = (float*)(smem + 49152);          // 32x1024 f32 : 128 KB
  float*    stg  = (float*)(smem + 180224);         // 32x1024 f32 : 128 KB

  const int dir = blockIdx.x;
  const _Float16* W = dir ? whh_b : whh_f;
  const float* G = dir ? Gb : Gf;
  float* Hout = dir ? Hb : Hf;
  const int tid = threadIdx.x;

  for (int i = tid; i < Bb * Hd; i += 1024) {
    h16[i] = (_Float16)h0[(size_t)dir * Bb * Hd + i];
    cst[i] = c0[(size_t)dir * Bb * Hd + i];
  }
  // preload G[t=0] into the LDS stage (8 x B128 per thread)
  for (int i = tid; i < (Bb * G4) / 4; i += 1024)
    async_copy_b128(stg + 4 * (size_t)i, G + 4 * (size_t)i);
  wait_asynccnt0();
  __syncthreads();

  const int wave = tid >> 5;
  const int lane = tid & 31;
  const int l = lane & 15, hi = lane >> 4;

  for (int t = 0; t < Tt; ++t) {
#pragma unroll
    for (int q = 0; q < 4; ++q) {
      const int tile = wave + 32 * q;          // 0..127
      const int tn = tile & 63, tm = tile >> 6;
      const int m0 = tm * 16, n0 = tn * 16;
      v8f acc;
#pragma unroll
      for (int r = 0; r < 8; ++r)              // init from LDS stage (= G[t])
        acc[r] = stg[(size_t)(m0 + r + hi * 8) * G4 + n0 + l];
#pragma unroll
      for (int k0 = 0; k0 < Hd; k0 += 32) {
        v16h a  = load_a_frag(h16, Hd, m0, k0, l, hi);
        v16h bf = load_b_frag(W, Hd, n0, k0, l, hi);
        acc = __builtin_amdgcn_wmma_f32_16x16x32_f16(false, a, false, bf,
                                                     (short)0, acc, false, false);
      }
#pragma unroll
      for (int r = 0; r < 8; ++r)
        gate[(size_t)(m0 + r + hi * 8) * G4 + n0 + l] = acc[r];
    }
    __syncthreads();   // gates complete; all stage reads for step t done

    // DMA next step's gate slab into the stage while we do the activation
    if (t + 1 < Tt) {
      const float* Gn = G + (size_t)(t + 1) * Bb * G4;
      for (int i = tid; i < (Bb * G4) / 4; i += 1024)
        async_copy_b128(stg + 4 * (size_t)i, Gn + 4 * (size_t)i);
    }

    for (int i = tid; i < Bb * Hd; i += 1024) {
      const int b_ = i >> 8, j = i & 255;
      const float gi = gate[(size_t)b_ * G4 + j];
      const float gf = gate[(size_t)b_ * G4 + 256 + j];
      const float gg = gate[(size_t)b_ * G4 + 512 + j];
      const float go = gate[(size_t)b_ * G4 + 768 + j];
      const float cn = sigf(gf) * cst[i] + sigf(gi) * tanhf(gg);
      const float hn = sigf(go) * tanhf(cn);
      cst[i] = cn;
      h16[i] = (_Float16)hn;
      Hout[(size_t)t * Bb * Hd + i] = hn;      // layout [t][b][j]
    }
    wait_asynccnt0();  // our async writes to stage have landed in LDS
    __syncthreads();   // everyone's writes visible; h16/cst ready
  }
}

// ---------------- kernel 5: output projection (K=12 is tiny) ---------------
__global__ void k_proj(const float* Hf, const float* Hb, const int* lens,
                       const float* Wout, const float* bout, float* feats) {
  int g = blockIdx.x * 256 + threadIdx.x;     // < B*T*16
  int k = g & 15;
  int bt = g >> 4;
  int t = bt & 255, b = bt >> 8;
  if (k >= Kk) return;
  int len = lens[b];
  int t2 = (t < len) ? (len - 1 - t) : t;     // un-reverse backward hiddens
  const float* wrow = Wout + (size_t)k * Hh;
  const float* hf = Hf + ((size_t)t  * Bb + b) * Hd;
  const float* hb = Hb + ((size_t)t2 * Bb + b) * Hd;
  float acc = bout[k];
#pragma unroll 8
  for (int j = 0; j < Hd; ++j) acc += hf[j] * wrow[j];
#pragma unroll 8
  for (int j = 0; j < Hd; ++j) acc += hb[j] * wrow[Hd + j];
  feats[((size_t)b * Tt + t) * Kk + k] = acc;
}

// ---------------- kernel 6: Viterbi decode + backtrack ---------------------
__global__ void k_viterbi(const float* feats, const int* lens, const float* trans,
                          int* bps, float* out) {
  int b = threadIdx.x;
  if (b >= Bb) return;
  float fv[Kk];
#pragma unroll
  for (int j = 0; j < Kk; ++j) fv[j] = (j == STARTT) ? 0.0f : NEGV;
  const int len = lens[b];
  for (int t = 0; t < Tt; ++t) {
    float nf[Kk];
#pragma unroll
    for (int nx = 0; nx < Kk; ++nx) {
      float best = fv[0] + trans[nx * Kk + 0];
      int bi = 0;
#pragma unroll
      for (int p = 1; p < Kk; ++p) {
        float s = fv[p] + trans[nx * Kk + p];
        if (s > best) { best = s; bi = p; }   // strict >: first-max like argmax
      }
      bps[((size_t)t * Bb + b) * Kk + nx] = bi;
      nf[nx] = best + feats[((size_t)b * Tt + t) * Kk + nx];
    }
    if (t < len) {
#pragma unroll
      for (int j = 0; j < Kk; ++j) fv[j] = nf[j];
    }
  }
  float best = fv[0] + trans[STOPP * Kk + 0];
  int bl = 0;
#pragma unroll
  for (int j = 1; j < Kk; ++j) {
    float s = fv[j] + trans[STOPP * Kk + j];
    if (s > best) { best = s; bl = j; }
  }
  out[b] = best;                               // path score
  int tag = bl;
  for (int t = Tt - 1; t >= 0; --t) {
    bool m = (t < len);
    out[Bb + (size_t)b * Tt + t] = m ? (float)tag : -1.0f;
    if (m) tag = bps[((size_t)t * Bb + b) * Kk + tag];
  }
}

// ---------------- host launcher --------------------------------------------
extern "C" void kernel_launch(void* const* d_in, const int* in_sizes, int n_in,
                              void* d_out, int out_size, void* d_ws, size_t ws_size,
                              hipStream_t stream) {
  (void)in_sizes; (void)n_in; (void)out_size; (void)ws_size;
  const int*   sentence = (const int*)  d_in[0];
  const int*   seq_lens = (const int*)  d_in[1];
  const float* emb      = (const float*)d_in[2];
  const float* Wih_f    = (const float*)d_in[3];
  const float* Whh_f    = (const float*)d_in[4];
  const float* bih_f    = (const float*)d_in[5];
  const float* bhh_f    = (const float*)d_in[6];
  const float* Wih_b    = (const float*)d_in[7];
  const float* Whh_b    = (const float*)d_in[8];
  const float* bih_b    = (const float*)d_in[9];
  const float* bhh_b    = (const float*)d_in[10];
  const float* h0       = (const float*)d_in[11];
  const float* c0       = (const float*)d_in[12];
  const float* W_out    = (const float*)d_in[13];
  const float* b_out    = (const float*)d_in[14];
  const float* trans    = (const float*)d_in[15];
  float* out = (float*)d_out;

  char* ws = (char*)d_ws;
  size_t off = 0;
  auto take = [&](size_t bytes) { char* p = ws + off; off += (bytes + 255) & ~(size_t)255; return p; };
  _Float16* xf16   = (_Float16*)take((size_t)Bb * Tt * Ee * 2);   // 4 MB
  _Float16* xr16   = (_Float16*)take((size_t)Bb * Tt * Ee * 2);   // 4 MB
  _Float16* wih16f = (_Float16*)take((size_t)G4 * Ee * 2);        // 512 KB
  _Float16* wih16b = (_Float16*)take((size_t)G4 * Ee * 2);
  _Float16* whh16f = (_Float16*)take((size_t)G4 * Hd * 2);
  _Float16* whh16b = (_Float16*)take((size_t)G4 * Hd * 2);
  float* Gf    = (float*)take((size_t)Tt * Bb * G4 * 4);          // 32 MB
  float* Gb    = (float*)take((size_t)Tt * Bb * G4 * 4);          // 32 MB
  float* Hf    = (float*)take((size_t)Tt * Bb * Hd * 4);          // 8 MB
  float* Hbrev = (float*)take((size_t)Tt * Bb * Hd * 4);          // 8 MB
  float* feats = (float*)take((size_t)Bb * Tt * Kk * 4);
  int*   bps   = (int*)  take((size_t)Tt * Bb * Kk * 4);

  // fp32 -> fp16 weight conversion
  k_convert_w<<<dim3(1024, 4), 256, 0, stream>>>(Wih_f, Whh_f, Wih_b, Whh_b,
                                                 wih16f, whh16f, wih16b, whh16b);
  // embedding gather (forward + reversed sequences)
  k_embed<<<8192, 256, 0, stream>>>(sentence, seq_lens, emb, xf16, xr16);
  // batched input projection GEMM via WMMA
  k_gemm_in<<<dim3(4096, 2), 256, 0, stream>>>(xf16, xr16, wih16f, wih16b,
                                               bih_f, bhh_f, bih_b, bhh_b, Gf, Gb);
  // persistent recurrence: 304 KB dynamic LDS (needs CDNA5's 320 KB WGP LDS)
  static_assert(16384 + 32768 + 131072 + 131072 == 311296, "lds layout");
  hipFuncSetAttribute((const void*)k_lstm,
                      hipFuncAttributeMaxDynamicSharedMemorySize, 311296);
  k_lstm<<<2, 1024, 311296, stream>>>(whh16f, whh16b, Gf, Gb, h0, c0, Hf, Hbrev);
  // output projection
  k_proj<<<512, 256, 0, stream>>>(Hf, Hbrev, seq_lens, W_out, b_out, feats);
  // Viterbi decode (scores -> out[0:32], preds -> out[32:32+B*T])
  k_viterbi<<<1, 32, 0, stream>>>(feats, seq_lens, trans, bps, out);
}
